// pMoE_644245095185
// MI455X (gfx1250) — matrix operand order repaired and checked by reading, not compiled, not run on hardware
//
#include <hip/hip_runtime.h>
#include <hip/hip_bf16.h>
#include <cstdint>
#include <cstddef>

#define TOK  4096
#define DIM  512
#define HID  2048
#define NEXP 16

#if defined(__gfx1250__) && __has_builtin(__builtin_amdgcn_global_load_async_to_lds_b128)
#define USE_ASYNC 1
#else
#define USE_ASYNC 0
#endif

typedef __attribute__((ext_vector_type(16))) __bf16 v16bf;
typedef __attribute__((ext_vector_type(8)))  float  v8f;

union U16F {
  v16bf v;
  uint4 q[2];
};

#if USE_ASYNC
typedef int v4i __attribute__((vector_size(16)));
typedef __attribute__((address_space(1))) v4i g1v4i;
typedef __attribute__((address_space(3))) v4i l3v4i;
__device__ __forceinline__ void async_copy16(const void* g, void* l) {
  __builtin_amdgcn_global_load_async_to_lds_b128((g1v4i*)g, (l3v4i*)l, 0, 0);
}
#endif

__device__ __forceinline__ void wait_async0() {
#if __has_builtin(__builtin_amdgcn_s_wait_asynccnt)
  __builtin_amdgcn_s_wait_asynccnt(0);
#else
  asm volatile("s_wait_asynccnt 0x0" ::: "memory");
#endif
}

__device__ __forceinline__ unsigned short f2bf(float f) {
  unsigned int u = __float_as_uint(f);
  u += 0x7FFFu + ((u >> 16) & 1u);        // round-to-nearest-even
  return (unsigned short)(u >> 16);
}

__device__ __forceinline__ unsigned int pack2bf(float a, float b) {
  return (unsigned int)f2bf(a) | ((unsigned int)f2bf(b) << 16);
}

// branchless tanh: (e^2u - 1)/(e^2u + 1), clamped so exp stays finite
__device__ __forceinline__ float tanh_fast(float u) {
  float uc = fminf(fmaxf(u, -9.0f), 9.0f);
  float ex = __expf(2.0f * uc);
  return (ex - 1.0f) * __builtin_amdgcn_rcpf(ex + 1.0f);
}

__device__ __forceinline__ float gelu_tanh(float v) {
  const float c0 = 0.7978845608028654f;   // sqrt(2/pi)
  float u = c0 * (v + 0.044715f * v * v * v);
  return 0.5f * v * (1.0f + tanh_fast(u));
}

// ---------------- x f32 -> bf16 (8 elems/thread) ----------------
__global__ __launch_bounds__(256) void convert_x_kernel(const float* __restrict__ x,
                                                        unsigned short* __restrict__ xb) {
  size_t i = ((size_t)blockIdx.x * 256 + threadIdx.x) * 8;
  float4 a = *(const float4*)(x + i);
  float4 b = *(const float4*)(x + i + 4);
  uint4 o = make_uint4(pack2bf(a.x, a.y), pack2bf(a.z, a.w),
                       pack2bf(b.x, b.y), pack2bf(b.z, b.w));
  *(uint4*)(xb + i) = o;
}

// ---------------- W f32 [E][R][C] -> bf16 transposed [E][C][R] ----------------
__global__ __launch_bounds__(256) void convert_w_t_kernel(const float* __restrict__ w,
                                                          unsigned short* __restrict__ wt,
                                                          int R, int C) {
  int e = blockIdx.z;
  const float* src = w + (size_t)e * R * C;
  unsigned short* dst = wt + (size_t)e * R * C;
  int c0 = blockIdx.x * 64, r0 = blockIdx.y * 64;
  __shared__ unsigned short tile[64 * 66];
  int tid = threadIdx.x;
  {
    int r = tid >> 2;
    int cc = (tid & 3) * 16;
    const float* s = src + (size_t)(r0 + r) * C + (c0 + cc);
    unsigned short* trow = &tile[r * 66 + cc];
#pragma unroll
    for (int j = 0; j < 16; j += 4) {
      float4 f = *(const float4*)(s + j);
      trow[j + 0] = f2bf(f.x); trow[j + 1] = f2bf(f.y);
      trow[j + 2] = f2bf(f.z); trow[j + 3] = f2bf(f.w);
    }
  }
  __syncthreads();
  {
    int c = tid >> 2;
    int rr = (tid & 3) * 16;
    unsigned int wv[8];
#pragma unroll
    for (int j = 0; j < 8; j++) {
      unsigned int lo = tile[(rr + 2 * j) * 66 + c];
      unsigned int hi = tile[(rr + 2 * j + 1) * 66 + c];
      wv[j] = lo | (hi << 16);
    }
    unsigned short* dp = dst + (size_t)(c0 + c) * R + r0 + rr;
    *(uint4*)(dp)     = make_uint4(wv[0], wv[1], wv[2], wv[3]);
    *(uint4*)(dp + 8) = make_uint4(wv[4], wv[5], wv[6], wv[7]);
  }
}

// ---------------- gate: logits, top-2, softmax, counts ----------------
__global__ __launch_bounds__(256) void gate_kernel(const float* __restrict__ x,
                                                   const float* __restrict__ Wg,
                                                   const float* __restrict__ bg,
                                                   int* __restrict__ gidx,
                                                   float* __restrict__ gscore,
                                                   int* __restrict__ counts) {
  int t = blockIdx.x * 256 + threadIdx.x;
  if (t >= TOK) return;
  float acc[NEXP];
#pragma unroll
  for (int e = 0; e < NEXP; e++) acc[e] = bg[e];
  const float* xr = x + (size_t)t * DIM;
  for (int d = 0; d < DIM; d++) {
    float xv = xr[d];
    const float* wr = Wg + (size_t)d * NEXP;
#pragma unroll
    for (int e = 0; e < NEXP; e++) acc[e] += xv * wr[e];
  }
  float v0 = -3.4e38f; int i0 = 0;
#pragma unroll
  for (int e = 0; e < NEXP; e++) if (acc[e] > v0) { v0 = acc[e]; i0 = e; }
  float v1 = -3.4e38f; int i1 = 0;
#pragma unroll
  for (int e = 0; e < NEXP; e++) if (e != i0 && acc[e] > v1) { v1 = acc[e]; i1 = e; }
  float e1 = __expf(v1 - v0);
  float p0 = 1.0f / (1.0f + e1);
  gidx[2 * t]       = i0;
  gidx[2 * t + 1]   = i1;
  gscore[2 * t]     = p0;
  gscore[2 * t + 1] = e1 * p0;
  atomicAdd(&counts[i0], 1);
  atomicAdd(&counts[i1], 1);
}

// ---------------- prefix over expert counts ----------------
__global__ void prefix_kernel(const int* __restrict__ counts, int* __restrict__ offsets) {
  if (threadIdx.x == 0 && blockIdx.x == 0) {
    int s = 0;
    for (int e = 0; e < NEXP; e++) { offsets[e] = s; s += counts[e]; }
    offsets[NEXP] = s;
  }
}

// ---------------- deterministic ordered routing lists ----------------
__global__ __launch_bounds__(256) void scan_kernel(const int* __restrict__ gidx,
                                                   const int* __restrict__ offsets,
                                                   int* __restrict__ rows_token,
                                                   int* __restrict__ row_of_slot) {
  int e = blockIdx.x;
  int tid = threadIdx.x;
  int lane = tid & 31, wid = tid >> 5;
  __shared__ int wave_sums[8];
  int base = offsets[e];
  for (int c = 0; c < TOK; c += 256) {
    int t = c + tid;
    int a0 = gidx[2 * t];
    int a1 = gidx[2 * t + 1];
    int flag = (a0 == e) || (a1 == e);
    int slot = (a0 == e) ? 0 : 1;
    unsigned int ball = (unsigned int)__ballot(flag);
    int pre = __popc(ball & ((1u << lane) - 1u));
    if (lane == 0) wave_sums[wid] = __popc(ball);
    __syncthreads();
    int wbase = 0, ctotal = 0;
#pragma unroll
    for (int w = 0; w < 8; w++) {
      int s = wave_sums[w];
      if (w < wid) wbase += s;
      ctotal += s;
    }
    if (flag) {
      int row = base + wbase + pre;
      rows_token[row] = t;
      row_of_slot[2 * t + slot] = row;
    }
    base += ctotal;
    __syncthreads();
  }
}

// ---------------- grouped GEMM via v_wmma_f32_16x16x32_bf16 ----------------
// Block: 256 thr = 8 waves (2 M-waves x 4 N-waves); block tile M=64, N=256.
// Wave tile 32x64: 2 A-fragments x 4 B-fragments = 8 wmma per K-step of 32.
// Double-buffered LDS, one barrier per K-step. Data movement: gfx1250 async
// global->LDS copies (ASYNCcnt) -> zero staging VGPRs, no spills; next K-tile
// is issued right after the barrier and overlaps the wmma chain.
template <int KTOT, int NTOT, bool GATHER_A, bool FUSE_GELU, bool B_PRET>
__global__ __launch_bounds__(256, 1) void moe_gemm(const unsigned short* __restrict__ Abase,
                                                   const void* __restrict__ Bmat,
                                                   const float* __restrict__ bias,
                                                   const int* __restrict__ offsets,
                                                   const int* __restrict__ rows_token,
                                                   unsigned short* __restrict__ HbufOut,
                                                   float* __restrict__ YbufOut) {
  int e = blockIdx.z;
  int m0 = offsets[e];
  int cnt = offsets[e + 1] - m0;
  int mbase = blockIdx.x * 64;
  if (mbase >= cnt) return;
  int n0 = blockIdx.y * 256;

  __shared__ unsigned short aLds[2][64 * 40];    // 2 x 5 KB
  __shared__ unsigned short bLds[2][256 * 40];   // 2 x 20 KB  (n-major = B^T)

  int tid = threadIdx.x;
  int lane = tid & 31;
  int half = lane >> 4;
  int wid = tid >> 5;
  int waveM = wid & 1;        // 2 M-waves, 32 rows each
  int waveN = wid >> 1;       // 4 N-waves, 64 cols each

  const unsigned short* Wt = nullptr;
  const float* We = nullptr;
  if constexpr (B_PRET) Wt = (const unsigned short*)Bmat + (size_t)e * KTOT * NTOT;
  else                  We = (const float*)Bmat + (size_t)e * KTOT * NTOT;

  v8f acc[2][4] = {};

  // A staging: 4 threads/row, 16B each; row clamped (garbage rows never stored)
  int ar = tid >> 2;
  int akk = (tid & 3) * 8;
  int grA = m0 + min(mbase + ar, cnt - 1);
  size_t asrc = GATHER_A ? (size_t)rows_token[grA] * KTOT : (size_t)grA * KTOT;

  auto computeTile = [&](int p) {
    U16F a0, a1;
    {
      const unsigned short* ap = &aLds[p][(waveM * 32 + (lane & 15)) * 40];
      a0.q[0] = *(const uint4*)(ap + half * 8);
      a0.q[1] = *(const uint4*)(ap + 16 + half * 8);
      ap += 16 * 40;
      a1.q[0] = *(const uint4*)(ap + half * 8);
      a1.q[1] = *(const uint4*)(ap + 16 + half * 8);
    }
#pragma unroll
    for (int t4 = 0; t4 < 4; t4++) {
      U16F b;
      const unsigned short* bp = &bLds[p][(waveN * 64 + t4 * 16 + (lane & 15)) * 40 + half * 8];
      b.q[0] = *(const uint4*)(bp);
      b.q[1] = *(const uint4*)(bp + 16);
      acc[0][t4] = __builtin_amdgcn_wmma_f32_16x16x32_bf16(false, a0.v, false, b.v,
                                                           (short)0, acc[0][t4], false, false);
      acc[1][t4] = __builtin_amdgcn_wmma_f32_16x16x32_bf16(false, a1.v, false, b.v,
                                                           (short)0, acc[1][t4], false, false);
    }
  };

#if USE_ASYNC
  if constexpr (B_PRET) {
    // ---- async global->LDS pipeline: no staging VGPRs at all ----
    auto issueStage = [&](int kk0, int pp) {
      async_copy16(Abase + asrc + kk0 + akk, &aLds[pp][ar * 40 + akk]);
      const unsigned short* bs = Wt + (size_t)(n0 + tid) * KTOT + kk0;  // 64B/thread
      unsigned short* bl = &bLds[pp][tid * 40];
#pragma unroll
      for (int j = 0; j < 4; j++) async_copy16(bs + j * 8, bl + j * 8);
    };
    issueStage(0, 0);
#pragma unroll 2
    for (int k0 = 0; k0 < KTOT; k0 += 32) {
      int p = (k0 >> 5) & 1;
      wait_async0();            // own async copies into buffer p done
      __syncthreads();          // everyone's copies into buffer p done
      if (k0 + 32 < KTOT) issueStage(k0 + 32, p ^ 1);   // overlaps wmma below
      computeTile(p);
    }
  } else
#endif
  {
    // ---- register-staged pipeline (fallback / on-the-fly f32 B) ----
    int bk = tid >> 3;
    int bn = (tid & 7) * 32;
    uint4 aReg;
    uint4 bR[4];
    float4 fR[8];
    auto loadStage = [&](int kk0) {
      aReg = *(const uint4*)(Abase + asrc + kk0 + akk);
      if constexpr (B_PRET) {
        const uint4* bs = (const uint4*)(Wt + (size_t)(n0 + tid) * KTOT + kk0);
#pragma unroll
        for (int j = 0; j < 4; j++) bR[j] = bs[j];
      } else {
        const float4* b4 = (const float4*)(We + (size_t)(kk0 + bk) * NTOT + (n0 + bn));
#pragma unroll
        for (int j = 0; j < 8; j++) fR[j] = b4[j];
      }
    };
    loadStage(0);
#pragma unroll 1
    for (int k0 = 0; k0 < KTOT; k0 += 32) {
      int p = (k0 >> 5) & 1;
      *(uint4*)(&aLds[p][ar * 40 + akk]) = aReg;
      if constexpr (B_PRET) {
#pragma unroll
        for (int j = 0; j < 4; j++) *(uint4*)(&bLds[p][tid * 40 + j * 8]) = bR[j];
      } else {
        const float* ff = (const float*)fR;
#pragma unroll
        for (int j = 0; j < 32; j++) bLds[p][(bn + j) * 40 + bk] = f2bf(ff[j]);
      }
      __syncthreads();
      if (k0 + 32 < KTOT) loadStage(k0 + 32);
      computeTile(p);
    }
  }

  // Epilogue: C/D layout -> VGPR i: M = i + (lane>=16 ? 8 : 0), N = lane&15
  int nc = lane & 15;
#pragma unroll
  for (int tm = 0; tm < 2; tm++) {
    int mloc = waveM * 32 + tm * 16 + half * 8;
#pragma unroll
    for (int t4 = 0; t4 < 4; t4++) {
      int col = n0 + waveN * 64 + t4 * 16 + nc;
      float bv = bias[(size_t)e * NTOT + col];
#pragma unroll
      for (int i = 0; i < 8; i++) {
        int lr = mbase + mloc + i;
        if (lr < cnt) {
          int gr = m0 + lr;
          float v = acc[tm][t4][i] + bv;
          if constexpr (FUSE_GELU) {
            HbufOut[(size_t)gr * NTOT + col] = f2bf(gelu_tanh(v));
          } else {
            YbufOut[(size_t)gr * NTOT + col] = v;
          }
        }
      }
    }
  }
}

// ---------------- deterministic 2-way combine (float4) ----------------
__global__ __launch_bounds__(256) void combine_kernel(const float* __restrict__ Ybuf,
                                                      const int* __restrict__ row_of_slot,
                                                      float* __restrict__ out) {
  int gid = blockIdx.x * 256 + threadIdx.x;      // over T*D/4
  int t = gid >> 7;                              // DIM/4 = 128
  int d4 = gid & 127;
  int r0 = row_of_slot[2 * t];
  int r1 = row_of_slot[2 * t + 1];
  const float4* y0 = (const float4*)(Ybuf + (size_t)r0 * DIM) + d4;
  const float4* y1 = (const float4*)(Ybuf + (size_t)r1 * DIM) + d4;
  float4 a = *y0, b = *y1;
  float4 o = make_float4(a.x + b.x, a.y + b.y, a.z + b.z, a.w + b.w);
  *((float4*)out + (size_t)gid) = o;
}

extern "C" void kernel_launch(void* const* d_in, const int* in_sizes, int n_in,
                              void* d_out, int out_size, void* d_ws, size_t ws_size,
                              hipStream_t stream) {
  (void)in_sizes; (void)n_in; (void)out_size;
  const float* x  = (const float*)d_in[0];
  const float* Wg = (const float*)d_in[1];
  const float* bg = (const float*)d_in[2];
  const float* W1 = (const float*)d_in[3];
  const float* b1 = (const float*)d_in[4];
  const float* W2 = (const float*)d_in[5];
  const float* b2 = (const float*)d_in[6];

  float* outf = (float*)d_out;
  int*   gidx   = (int*)(outf + (size_t)TOK * DIM);
  float* gscore = outf + (size_t)TOK * DIM + (size_t)TOK * 2;

  char* ws = (char*)d_ws;
  size_t off = 0;
  auto take = [&](size_t bytes) -> char* {
    size_t o = off;
    off = (o + bytes + 255) & ~(size_t)255;
    return ws + o;
  };
  unsigned short* xb          = (unsigned short*)take((size_t)TOK * DIM * 2);     // 4 MB
  int*            counts      = (int*)take(NEXP * sizeof(int));
  int*            offsets     = (int*)take((NEXP + 1) * sizeof(int));
  int*            rows_token  = (int*)take((size_t)TOK * 2 * sizeof(int));
  int*            row_of_slot = (int*)take((size_t)TOK * 2 * sizeof(int));
  unsigned short* Hbuf        = (unsigned short*)take((size_t)TOK * 2 * HID * 2); // 32 MB
  float*          Ybuf        = (float*)take((size_t)TOK * 2 * DIM * 4);          // 16 MB
  unsigned short* W1t = (unsigned short*)take((size_t)NEXP * DIM * HID * 2);      // 33.5 MB
  unsigned short* W2t = (unsigned short*)take((size_t)NEXP * HID * DIM * 2);      // 33.5 MB
  bool big = (off <= ws_size);

  (void)hipMemsetAsync(counts, 0, NEXP * sizeof(int), stream);

  convert_x_kernel<<<(TOK * DIM) / (256 * 8), 256, 0, stream>>>(x, xb);
  gate_kernel<<<TOK / 256, 256, 0, stream>>>(x, Wg, bg, gidx, gscore, counts);
  prefix_kernel<<<1, 1, 0, stream>>>(counts, offsets);
  scan_kernel<<<NEXP, 256, 0, stream>>>(gidx, offsets, rows_token, row_of_slot);

  if (big) {
    // one-time convert + transpose: W1 [E][D][H] -> [E][H][D] bf16, W2 [E][H][D] -> [E][D][H] bf16
    convert_w_t_kernel<<<dim3(HID / 64, DIM / 64, NEXP), 256, 0, stream>>>(W1, W1t, DIM, HID);
    convert_w_t_kernel<<<dim3(DIM / 64, HID / 64, NEXP), 256, 0, stream>>>(W2, W2t, HID, DIM);

    moe_gemm<DIM, HID, true, true, true>
        <<<dim3(TOK / 64, HID / 256, NEXP), 256, 0, stream>>>(
            xb, W1t, b1, offsets, rows_token, Hbuf, nullptr);
    moe_gemm<HID, DIM, false, false, true>
        <<<dim3(TOK / 64, DIM / 256, NEXP), 256, 0, stream>>>(
            Hbuf, W2t, b2, offsets, rows_token, nullptr, Ybuf);
  } else {
    moe_gemm<DIM, HID, true, true, false>
        <<<dim3(TOK / 64, HID / 256, NEXP), 256, 0, stream>>>(
            xb, W1, b1, offsets, rows_token, Hbuf, nullptr);
    moe_gemm<HID, DIM, false, false, false>
        <<<dim3(TOK / 64, DIM / 256, NEXP), 256, 0, stream>>>(
            Hbuf, W2, b2, offsets, rows_token, nullptr, Ybuf);
  }

  combine_kernel<<<(TOK * DIM) / (256 * 4), 256, 0, stream>>>(Ybuf, row_of_slot, outf);
}
